// UniversalTransformerEncoder_85160611545802
// MI455X (gfx1250) — matrix operand order repaired and checked by reading, not compiled, not run on hardware
//
#include <hip/hip_runtime.h>
#include <hip/hip_bf16.h>

// ---------------------------------------------------------------------------
// Universal Transformer Encoder (ACT) for MI455X / gfx1250, wave32 + WMMA.
// B=2, S=2048, D=512, H=8, DH=64, MAX_STEPS=8, THRESH=0.99
// GEMM weight panels are double-buffered in LDS via the Tensor Data Mover:
// the TDM DMA of panel p+1 overlaps the 8 WMMAs consuming panel p.
// ---------------------------------------------------------------------------

typedef __attribute__((ext_vector_type(16))) _Float16 v16h;
typedef __attribute__((ext_vector_type(8)))  _Float16 h8;
typedef __attribute__((ext_vector_type(8)))  float    v8f;
typedef __attribute__((ext_vector_type(4)))  unsigned int u32x4;
typedef __attribute__((ext_vector_type(8)))  unsigned int u32x8;

#define BN   2
#define SL   2048
#define DM   512
#define NH   8
#define DH   64
#define NTOK (BN*SL)          // 4096
#define NEL  (NTOK*DM)        // 2,097,152
#define WEL  (DM*DM)          // 262,144

// ---------------------------------------------------------------------------
// WMMA helpers (CDNA5 register layouts, 16x16x32 f16 -> f32)
// ---------------------------------------------------------------------------
__device__ __forceinline__ v8f wmma_f16(v16h a, v16h b, v8f c) {
  return __builtin_amdgcn_wmma_f32_16x16x32_f16(
      /*neg_a=*/false, a, /*neg_b=*/false, b,
      /*c_mod=*/(short)0, c, /*reuse_a=*/false, /*reuse_b=*/false);
}

// A-matrix 16x32 f16 tile. src points at tile origin, row-major, leading dim ld.
// ISA layout: lane<16 -> row=lane,    K-halves {0..7} then {16..23}
//             lane>=16 -> row=lane-16, K-halves {8..15} then {24..31}
__device__ __forceinline__ v16h load_A_tile(const _Float16* src, int ld, int lane) {
  int m  = lane & 15;
  int kb = (lane >> 4) * 8;
  h8 lo = *(const h8*)(src + (size_t)m * ld + kb);
  h8 hi = *(const h8*)(src + (size_t)m * ld + kb + 16);
  v16h a;
#pragma unroll
  for (int i = 0; i < 8; ++i) { a[i] = lo[i]; a[i + 8] = hi[i]; }
  return a;
}

// B-matrix 32x16 f16 tile with B[k,n] = src[n*ld + k] (i.e. src holds the
// transposed operand row-major, each lane reads two 16-byte chunks, k0 and k0+16).
// ISA layout: lane<16 -> n=lane,    k {0..7} then {16..23} ; lane>=16 -> n=lane-16, k {8..15} then {24..31}
__device__ __forceinline__ v16h load_B_tile(const _Float16* src, int ld, int lane) {
  int n  = lane & 15;
  int k0 = (lane >> 4) * 8;
  h8 lo = *(const h8*)(src + (size_t)n * ld + k0);
  h8 hi = *(const h8*)(src + (size_t)n * ld + k0 + 16);
  v16h b;
#pragma unroll
  for (int i = 0; i < 8; ++i) { b[i] = lo[i]; b[i + 8] = hi[i]; }
  return b;
}

__device__ __forceinline__ v8f zero_v8f() {
  v8f z;
#pragma unroll
  for (int i = 0; i < 8; ++i) z[i] = 0.0f;
  return z;
}

// ---------------------------------------------------------------------------
// TDM: DMA a 2D f16 tile (tile_dim0 x tile_dim1, row stride `stride` halfs)
// from global memory into LDS at byte offset ldsOff. Issued per-wave (EXEC
// ignored by TDM). Descriptor layout per CDNA5 ISA 8.3/8.4 (D# groups 0/1).
// ---------------------------------------------------------------------------
__device__ __forceinline__ void tdm_load_tile_f16(const _Float16* gsrc,
                                                  unsigned int ldsOff,
                                                  unsigned int tileX,
                                                  unsigned int tileY,
                                                  unsigned int stride) {
  unsigned long long ga = (unsigned long long)(uintptr_t)gsrc;
  u32x4 g0;
  g0[0] = 1u;                                     // count=1, user mode
  g0[1] = ldsOff;                                 // lds_addr [63:32]
  g0[2] = (unsigned int)ga;                       // global_addr [95:64]
  g0[3] = ((unsigned int)(ga >> 32) & 0x01FFFFFFu) | 0x80000000u;  // type=2
  u32x8 g1;
  g1[0] = 1u << 16;                               // data_size = 2 bytes
  g1[1] = (stride & 0xFFFFu) << 16;               // tensor_dim0 low16
  g1[2] = (stride >> 16) | ((512u & 0xFFFFu) << 16);  // dim0 hi / tensor_dim1 lo
  g1[3] = (tileX & 0xFFFFu) << 16;                // tile_dim0
  g1[4] = tileY & 0xFFFFu;                        // tile_dim1 (tile_dim2=0)
  g1[5] = stride;                                 // tensor_dim0_stride low32
  g1[6] = 0u;                                     // stride hi / dim1_stride lo
  g1[7] = 0u;
  asm volatile("tensor_load_to_lds %0, %1" :: "s"(g0), "s"(g1) : "memory");
}

// ---------------------------------------------------------------------------
// Init: zero output + halting state, copy state, convert weights to f16
// ---------------------------------------------------------------------------
__global__ void __launch_bounds__(256)
init_kernel(const float* __restrict__ inx,
            const float* __restrict__ wq, const float* __restrict__ wk,
            const float* __restrict__ wv, const float* __restrict__ w1,
            const float* __restrict__ w2,
            float* __restrict__ state, float* __restrict__ dout,
            _Float16* __restrict__ wqh, _Float16* __restrict__ wkh,
            _Float16* __restrict__ wvh, _Float16* __restrict__ w1h,
            _Float16* __restrict__ w2h,
            float* __restrict__ halt_p, float* __restrict__ rems) {
  for (int i = blockIdx.x * blockDim.x + threadIdx.x; i < NEL;
       i += gridDim.x * blockDim.x) {
    state[i] = inx[i];
    dout[i]  = 0.0f;
    if (i < WEL) {
      wqh[i] = (_Float16)wq[i];
      wkh[i] = (_Float16)wk[i];
      wvh[i] = (_Float16)wv[i];
      w1h[i] = (_Float16)w1[i];
      w2h[i] = (_Float16)w2[i];
    }
    if (i < NTOK) { halt_p[i] = 0.0f; rems[i] = 0.0f; }
  }
}

// ---------------------------------------------------------------------------
// Per-step: halting probability + ACT bookkeeping + X = state + P_t
// One block (256 threads) per token.
// ---------------------------------------------------------------------------
__global__ void __launch_bounds__(256)
halt_x_kernel(const float* __restrict__ state, const float* __restrict__ hw,
              const float* __restrict__ hb, float* __restrict__ halt_p,
              float* __restrict__ rems, float* __restrict__ uw,
              float* __restrict__ X, _Float16* __restrict__ Xh, float tstep) {
  __shared__ float red[256];
  const int tok = blockIdx.x;
  const int tid = threadIdx.x;
  const size_t base = (size_t)tok * DM;

  float acc = state[base + tid] * hw[tid] + state[base + tid + 256] * hw[tid + 256];
  red[tid] = acc;
  __syncthreads();
#pragma unroll
  for (int s = 128; s > 0; s >>= 1) {
    if (tid < s) red[tid] += red[tid + s];
    __syncthreads();
  }
  if (tid == 0) {
    float p  = 1.0f / (1.0f + __expf(-(red[0] + hb[0])));
    float hp = halt_p[tok];
    float rm = rems[tok];
    float still = (hp < 1.0f) ? 1.0f : 0.0f;
    float sum = hp + p * still;
    float nh  = ((sum > 0.99f) ? 1.0f : 0.0f) * still;
    float sn  = ((sum <= 0.99f) ? 1.0f : 0.0f) * still;
    float rem = 1.0f - hp - p * still;
    float u   = p * sn + nh * rem;
    hp = hp + p * still;
    rm = rm + nh * rem;
    hp = hp + nh * rm;
    halt_p[tok] = hp;
    rems[tok]   = rm;
    uw[tok]     = u;
  }
  // coord encoding P_t (Eqs 6-7) + residual into X
  const int s_pos = tok & (SL - 1);
#pragma unroll
  for (int r = 0; r < 2; ++r) {
    int d = tid + r * 256;
    float fd    = (float)(d & ~1);
    float denom = __powf(10000.0f, fd * (1.0f / (float)DM));
    float pa = (float)s_pos / denom;
    float ta = tstep / denom;
    float pe = ((d & 1) == 0) ? (__sinf(pa) + __sinf(ta))
                              : (__cosf(pa) + __cosf(ta));
    float xv = state[base + d] + pe;
    X[base + d]  = xv;
    Xh[base + d] = (_Float16)xv;
  }
}

// ---------------------------------------------------------------------------
// WMMA GEMM: C[M,N] = A[M,K] * W^T  (W is [N,K] row-major, torch layout).
// Workgroup = 8 waves covering a 128x64 C tile sharing one weight panel.
// Double-buffered TDM pipeline: while the 8 waves run the 8 WMMAs of panel p
// (64x64-half panel in LDS), wave 0 has already issued tensor_load_to_lds for
// panel p+1 into the other buffer. One barrier per panel:
//   wait TENSORcnt(0) -> barrier -> issue TDM(p+1) -> compute(p)
// The barrier both publishes panel p and proves the p+1 buffer is reusable.
// ---------------------------------------------------------------------------
__global__ void __launch_bounds__(256)
gemm_wmma(const _Float16* __restrict__ A, const _Float16* __restrict__ W,
          const float* __restrict__ bias, int M, int N, int K,
          const float* __restrict__ resid, float* __restrict__ outF,
          _Float16* __restrict__ outH, const float* __restrict__ uw,
          float* __restrict__ accOut, int relu) {
  __shared__ __align__(16) _Float16 wlds[2][64 * 64];  // 2 x (64 cols x 64 k)
  const int lane = threadIdx.x & 31;
  const int wv   = threadIdx.x >> 5;
  const int ntiles = N >> 6;
  const int nt = (blockIdx.x % ntiles) << 6;
  const int mt = (blockIdx.x / ntiles) * 128 + wv * 16;

  unsigned int ldsOff[2];
  ldsOff[0] = (unsigned int)(uintptr_t)(&wlds[0][0]);
  ldsOff[1] = (unsigned int)(uintptr_t)(&wlds[1][0]);

  v8f acc[4];
#pragma unroll
  for (int t = 0; t < 4; ++t) acc[t] = zero_v8f();

  const int npanels = K >> 6;  // 64-half K panels
  if (wv == 0)
    tdm_load_tile_f16(W + (size_t)nt * K, ldsOff[0], 64, 64, (unsigned)K);

  for (int p = 0; p < npanels; ++p) {
    if (wv == 0) __builtin_amdgcn_s_wait_tensorcnt(0);  // panel p landed
    __syncthreads();  // publish panel p; buffer for p+1 is no longer read
    if (wv == 0 && p + 1 < npanels)
      tdm_load_tile_f16(W + (size_t)nt * K + (p + 1) * 64, ldsOff[(p + 1) & 1],
                        64, 64, (unsigned)K);          // overlaps compute

    if (p + 1 < npanels)  // warm L2 for the next A panel
      __builtin_prefetch(A + (size_t)mt * K + (p + 1) * 64, 0, 3);

    const _Float16* panel = &wlds[p & 1][0];
    const int kbase = p << 6;
#pragma unroll
    for (int kc = 0; kc < 64; kc += 32) {
      v16h a = load_A_tile(A + (size_t)mt * K + kbase + kc, K, lane);
#pragma unroll
      for (int t = 0; t < 4; ++t) {
        v16h b = load_B_tile(panel + (size_t)(t * 16) * 64 + kc, 64, lane);
        acc[t] = wmma_f16(a, b, acc[t]);
      }
    }
  }

  const int hi = lane >> 4;
  const int nlo = lane & 15;
#pragma unroll
  for (int t = 0; t < 4; ++t) {
    int col = nt + t * 16 + nlo;
    float bv = bias ? bias[col] : 0.0f;
#pragma unroll
    for (int j = 0; j < 8; ++j) {
      int r = mt + j + 8 * hi;
      float v = acc[t][j] + bv;
      if (relu) v = v > 0.0f ? v : 0.0f;
      size_t idx = (size_t)r * N + col;
      if (resid) v += resid[idx];
      if (outF) outF[idx] = v;
      if (outH) outH[idx] = (_Float16)v;
      if (accOut) accOut[idx] += uw[r] * v;
    }
  }
}

// ---------------------------------------------------------------------------
// Materialize the reference's "reshape-not-transpose" K operand and V^T:
//   K2T[b,h,kk,e] = K[b, e*32 + (kk>>6), h*64 + (kk&63)]   (ld 64,  B-operand)
//   Vt [b,h,e,kk] = V[b, kk, h*64 + e]                     (ld 2048, B-operand)
// ---------------------------------------------------------------------------
__global__ void __launch_bounds__(256)
reshape_kv(const _Float16* __restrict__ Kh, const _Float16* __restrict__ Vh,
           _Float16* __restrict__ K2T, _Float16* __restrict__ Vt) {
  const int total = BN * NH * SL * DH;  // 2^21
  for (int idx = blockIdx.x * blockDim.x + threadIdx.x; idx < total;
       idx += gridDim.x * blockDim.x) {
    {
      int e  = idx & 63;
      int kk = (idx >> 6) & (SL - 1);
      int h  = (idx >> 17) & 7;
      int b  = (idx >> 20) & 1;
      int sp = e * 32 + (kk >> 6);
      int dh = kk & 63;
      K2T[idx] = Kh[((size_t)(b * SL + sp)) * DM + h * DH + dh];
    }
    {
      int kk = idx & (SL - 1);
      int e  = (idx >> 11) & 63;
      int h  = (idx >> 17) & 7;
      int b  = (idx >> 20) & 1;
      Vt[idx] = Vh[((size_t)(b * SL + kk)) * DM + h * DH + e];
    }
  }
}

// ---------------------------------------------------------------------------
// Attention: one wave per (b, h, 16-row q-block). Streaming (online) softmax
// over the full 2048-key row; 8 WMMAs per 32-key step. P tiles go D-layout ->
// LDS -> A-layout for the attn*V WMMAs. Writes A = X + concat-head output.
// ---------------------------------------------------------------------------
__global__ void __launch_bounds__(256)
attn_kernel(const _Float16* __restrict__ Qh, const _Float16* __restrict__ K2T,
            const _Float16* __restrict__ Vt, const float* __restrict__ X,
            float* __restrict__ A, _Float16* __restrict__ Ah) {
  __shared__ __align__(16) _Float16 plds[8][16 * 32];
  const int lane = threadIdx.x & 31;
  const int wv   = threadIdx.x >> 5;
  const int gwave = blockIdx.x * 8 + wv;
  const int q0 = (gwave & 127) << 4;
  const int bh = gwave >> 7;
  const int b = bh >> 3, h = bh & 7;
  const int hi = lane >> 4, nlo = lane & 15;

  // Q A-operands (rows q0..q0+15, e = 0..63), held in registers for all keys
  const _Float16* qbase = Qh + ((size_t)(b * SL + q0)) * DM + h * DH;
  v16h aq0 = load_A_tile(qbase, DM, lane);       // e in [0,32)
  v16h aq1 = load_A_tile(qbase + 32, DM, lane);  // e in [32,64)

  const _Float16* kbase = K2T + (size_t)bh * SL * DH;  // [kk][e], ld 64
  const _Float16* vbase = Vt + (size_t)bh * DH * SL;   // [e][kk], ld 2048

  v8f acc[4];
#pragma unroll
  for (int t = 0; t < 4; ++t) acc[t] = zero_v8f();
  float m8[8], l8[8];
#pragma unroll
  for (int j = 0; j < 8; ++j) { m8[j] = -3.0e38f; l8[j] = 0.0f; }

  for (int kk0 = 0; kk0 < SL; kk0 += 32) {
    // ---- scores: two 16x16 tiles (keys kk0..+15, kk0+16..+31), K=64 over e
    v8f s0 = zero_v8f(), s1 = zero_v8f();
    {
      v16h b00 = load_B_tile(kbase + (size_t)kk0 * DH, DH, lane);
      v16h b01 = load_B_tile(kbase + (size_t)kk0 * DH + 32, DH, lane);
      s0 = wmma_f16(aq0, b00, s0);
      s0 = wmma_f16(aq1, b01, s0);
      v16h b10 = load_B_tile(kbase + (size_t)(kk0 + 16) * DH, DH, lane);
      v16h b11 = load_B_tile(kbase + (size_t)(kk0 + 16) * DH + 32, DH, lane);
      s1 = wmma_f16(aq0, b10, s1);
      s1 = wmma_f16(aq1, b11, s1);
    }
    // ---- online softmax (row stats replicated across the 16-lane half)
    const float sc = 0.125f;  // 1/sqrt(64)
    float rsc[8];
#pragma unroll
    for (int j = 0; j < 8; ++j) {
      float x0 = s0[j] * sc, x1 = s1[j] * sc;
      float tm = fmaxf(x0, x1);
      tm = fmaxf(tm, __shfl_xor(tm, 1));
      tm = fmaxf(tm, __shfl_xor(tm, 2));
      tm = fmaxf(tm, __shfl_xor(tm, 4));
      tm = fmaxf(tm, __shfl_xor(tm, 8));
      float newm = fmaxf(m8[j], tm);
      float p0 = __expf(x0 - newm);
      float p1 = __expf(x1 - newm);
      float ps = p0 + p1;
      ps += __shfl_xor(ps, 1);
      ps += __shfl_xor(ps, 2);
      ps += __shfl_xor(ps, 4);
      ps += __shfl_xor(ps, 8);
      float esc = __expf(m8[j] - newm);
      l8[j] = l8[j] * esc + ps;
      m8[j] = newm;
      rsc[j] = esc;
      s0[j] = p0;  // reuse score regs for probabilities
      s1[j] = p1;
    }
#pragma unroll
    for (int t = 0; t < 4; ++t)
#pragma unroll
      for (int j = 0; j < 8; ++j) acc[t][j] *= rsc[j];

    // ---- P tile: D-layout f32 -> LDS f16 -> A-layout (16x32)
    _Float16* lp = plds[wv];
#pragma unroll
    for (int j = 0; j < 8; ++j) {
      int r = j + 8 * hi;
      lp[r * 32 + nlo]      = (_Float16)s0[j];
      lp[r * 32 + 16 + nlo] = (_Float16)s1[j];
    }
    asm volatile("s_wait_dscnt 0" ::: "memory");
    v16h ap = load_A_tile(lp, 32, lane);

    // ---- out += P * V   (4 tiles across e, K=32 over keys)
#pragma unroll
    for (int t = 0; t < 4; ++t) {
      v16h bvt = load_B_tile(vbase + (size_t)(t * 16) * SL + kk0, SL, lane);
      acc[t] = wmma_f16(ap, bvt, acc[t]);
    }
  }

  // ---- epilogue: normalize, add residual X, store f32 + f16
#pragma unroll
  for (int t = 0; t < 4; ++t) {
    int col = h * DH + t * 16 + nlo;
#pragma unroll
    for (int j = 0; j < 8; ++j) {
      int r = q0 + j + 8 * hi;
      size_t idx = ((size_t)(b * SL + r)) * DM + col;
      float o = X[idx] + acc[t][j] / l8[j];
      A[idx]  = o;
      Ah[idx] = (_Float16)o;
    }
  }
}

// ---------------------------------------------------------------------------
// Host orchestration
// ---------------------------------------------------------------------------
extern "C" void kernel_launch(void* const* d_in, const int* in_sizes, int n_in,
                              void* d_out, int out_size, void* d_ws, size_t ws_size,
                              hipStream_t stream) {
  const float* inx = (const float*)d_in[0];
  const float* wq  = (const float*)d_in[1];
  const float* wk  = (const float*)d_in[2];
  const float* wv  = (const float*)d_in[3];
  const float* w1  = (const float*)d_in[4];
  const float* b1  = (const float*)d_in[5];
  const float* w2  = (const float*)d_in[6];
  const float* b2  = (const float*)d_in[7];
  const float* hw  = (const float*)d_in[8];
  const float* hb  = (const float*)d_in[9];
  float* out = (float*)d_out;

  // workspace carve-up (256B aligned)
  char* ws = (char*)d_ws;
  size_t off = 0;
  auto carve = [&](size_t bytes) -> void* {
    void* p = ws + off;
    off += (bytes + 255) & ~(size_t)255;
    return p;
  };
  float*    state  = (float*)   carve((size_t)NEL * 4);
  float*    X      = (float*)   carve((size_t)NEL * 4);
  float*    Abuf   = (float*)   carve((size_t)NEL * 4);
  _Float16* Xh     = (_Float16*)carve((size_t)NEL * 2);
  _Float16* Ahh    = (_Float16*)carve((size_t)NEL * 2);
  _Float16* Qh     = (_Float16*)carve((size_t)NEL * 2);
  _Float16* Kh     = (_Float16*)carve((size_t)NEL * 2);
  _Float16* Vh     = (_Float16*)carve((size_t)NEL * 2);
  _Float16* K2T    = (_Float16*)carve((size_t)NEL * 2);
  _Float16* Vt     = (_Float16*)carve((size_t)NEL * 2);
  _Float16* Hid    = (_Float16*)carve((size_t)NEL * 2);
  _Float16* wqh    = (_Float16*)carve((size_t)WEL * 2);
  _Float16* wkh    = (_Float16*)carve((size_t)WEL * 2);
  _Float16* wvh    = (_Float16*)carve((size_t)WEL * 2);
  _Float16* w1h    = (_Float16*)carve((size_t)WEL * 2);
  _Float16* w2h    = (_Float16*)carve((size_t)WEL * 2);
  float*    halt_p = (float*)   carve((size_t)NTOK * 4);
  float*    rems   = (float*)   carve((size_t)NTOK * 4);
  float*    uw     = (float*)   carve((size_t)NTOK * 4);

  init_kernel<<<2048, 256, 0, stream>>>(inx, wq, wk, wv, w1, w2, state, out,
                                        wqh, wkh, wvh, w1h, w2h, halt_p, rems);

  const int gemmGrid = (NTOK / 128) * (DM / 64);  // 32 * 8 = 256 blocks
  for (int t = 1; t <= 8; ++t) {
    halt_x_kernel<<<NTOK, 256, 0, stream>>>(state, hw, hb, halt_p, rems, uw,
                                            X, Xh, (float)t);
    // Q, K, V projections
    gemm_wmma<<<gemmGrid, 256, 0, stream>>>(Xh, wqh, nullptr, NTOK, DM, DM,
                                            nullptr, nullptr, Qh, nullptr,
                                            nullptr, 0);
    gemm_wmma<<<gemmGrid, 256, 0, stream>>>(Xh, wkh, nullptr, NTOK, DM, DM,
                                            nullptr, nullptr, Kh, nullptr,
                                            nullptr, 0);
    gemm_wmma<<<gemmGrid, 256, 0, stream>>>(Xh, wvh, nullptr, NTOK, DM, DM,
                                            nullptr, nullptr, Vh, nullptr,
                                            nullptr, 0);
    reshape_kv<<<2048, 256, 0, stream>>>(Kh, Vh, K2T, Vt);
    attn_kernel<<<256, 256, 0, stream>>>(Qh, K2T, Vt, X, Abuf, Ahh);
    // FFN: Hid = relu(A @ w1^T + b1)
    gemm_wmma<<<gemmGrid, 256, 0, stream>>>(Ahh, w1h, b1, NTOK, DM, DM,
                                            nullptr, nullptr, Hid, nullptr,
                                            nullptr, 1);
    // Ht = A + Hid @ w2^T + b2 ; state <- Ht ; acc += uw * Ht
    gemm_wmma<<<gemmGrid, 256, 0, stream>>>(Hid, w2h, b2, NTOK, DM, DM,
                                            Abuf, state, nullptr, uw, out, 0);
  }
}
